// AdaptiveFusion_83760452207312
// MI455X (gfx1250) — matrix operand order repaired and checked by reading, not compiled, not run on hardware
//
#include <hip/hip_runtime.h>
#include <hip/hip_bf16.h>

typedef __attribute__((ext_vector_type(4)))  float     f32x4;
typedef __attribute__((ext_vector_type(4)))  _Float16  h16x4;
typedef __attribute__((ext_vector_type(8)))  _Float16  h16x8;
typedef __attribute__((ext_vector_type(16))) _Float16  v16h;
typedef __attribute__((ext_vector_type(8)))  float     v8f;
typedef __attribute__((ext_vector_type(4)))  int       v4i;

#if defined(__has_builtin)
#if __has_builtin(__builtin_amdgcn_global_load_async_to_lds_b128) && \
    __has_builtin(__builtin_amdgcn_s_wait_asynccnt)
#define USE_ASYNC_LDS 1
#endif
#endif

namespace {
constexpr int kB = 16384;   // batch
constexpr int kS = 1024;    // spatial dim (= temporal dim)
constexpr int kF = 2048;    // expert output dim
constexpr int kE = 8;       // experts
constexpr int BM = 128, BN = 256, BK = 64;
constexpr int LDA = BK + 8; // halves, padded: row stride 144B -> conflict-free b128
constexpr int LDB = BK + 8;
}

static __device__ inline v16h cat16(h16x8 lo, h16x8 hi) {
  v16h r;
#pragma unroll
  for (int i = 0; i < 8; ++i) { r[i] = lo[i]; r[i + 8] = hi[i]; }
  return r;
}

// ---------------------------------------------------------------------------
// f32 -> f16 bulk converter (fills workspace staging buffers once per launch)
// ---------------------------------------------------------------------------
__global__ __launch_bounds__(256) void cvt_f16_kernel(
    const float* __restrict__ src, _Float16* __restrict__ dst, int n4) {
  int i = blockIdx.x * 256 + threadIdx.x;
  if (i < n4) {
    f32x4 v = *(const f32x4*)(src + (size_t)i * 4);
    h16x4 h;
#pragma unroll
    for (int j = 0; j < 4; ++j) h[j] = (_Float16)v[j];
    *(h16x4*)(dst + (size_t)i * 4) = h;
  }
}

// ---------------------------------------------------------------------------
// Kernel 1: gating network. h = relu([sp,tp] @ Wg1^T + bg1) via WMMA f16,
// z = h @ Wg2^T + bg2 (tiny, scalar), softmax -> gates (B x 16 f32 in ws).
// ---------------------------------------------------------------------------
__global__ __launch_bounds__(256) void gates_kernel(
    const float* __restrict__ sp, const float* __restrict__ tp,
    const float* __restrict__ Wg1, const float* __restrict__ bg1,
    const float* __restrict__ Wg2, const float* __restrict__ bg2,
    float* __restrict__ gates)
{
  __shared__ float hbuf[8][16][17];
  __shared__ float zbuf[8][16][17];
  const int t = threadIdx.x, lane = t & 31, wid = t >> 5;
  const int rowbase = blockIdx.x * 128 + wid * 16;
  const int mrow = lane & 15;
  const int lhi  = lane >> 4;

  v8f hacc = {};
#pragma unroll 4
  for (int k0 = 0; k0 < 2 * kS; k0 += 32) {
    const float* X = (k0 < kS) ? sp : tp;
    const int kk = (k0 < kS) ? k0 : (k0 - kS);
    const float* pa = X + (size_t)(rowbase + mrow) * kS + kk + lhi * 8;
    f32x4 a0 = *(const f32x4*)(pa);
    f32x4 a1 = *(const f32x4*)(pa + 4);
    f32x4 a2 = *(const f32x4*)(pa + 16);
    f32x4 a3 = *(const f32x4*)(pa + 20);
    v16h af;
#pragma unroll
    for (int i = 0; i < 4; ++i) {
      af[i]      = (_Float16)a0[i];
      af[i + 4]  = (_Float16)a1[i];
      af[i + 8]  = (_Float16)a2[i];
      af[i + 12] = (_Float16)a3[i];
    }
    const float* pb = Wg1 + (size_t)mrow * (2 * kS) + k0 + lhi * 8;
    v16h bf;
#pragma unroll
    for (int i = 0; i < 4; ++i) {
      f32x4 b = *(const f32x4*)(pb + 4 * i + (i & 2) * 4);
#pragma unroll
      for (int j = 0; j < 4; ++j) bf[4 * i + j] = (_Float16)b[j];
    }
    hacc = __builtin_amdgcn_wmma_f32_16x16x32_f16(false, af, false, bf,
                                                  (short)0, hacc, false, false);
  }
  const float b1 = bg1[mrow];
#pragma unroll
  for (int v = 0; v < 8; ++v) {
    float h = hacc[v] + b1;
    hbuf[wid][v + lhi * 8][mrow] = fmaxf(h, 0.0f);
  }
  __syncthreads();
  {
    const int m = mrow;
#pragma unroll
    for (int jj = 0; jj < 8; ++jj) {
      const int j = lhi * 8 + jj;
      float z = bg2[j];
#pragma unroll
      for (int j1 = 0; j1 < 16; ++j1) z += hbuf[wid][m][j1] * Wg2[j * 16 + j1];
      zbuf[wid][m][j] = z;
    }
  }
  __syncthreads();
  if (lane < 16) {
    const int m = lane;
    float zz[16], mx = -3.0e38f;
#pragma unroll
    for (int j = 0; j < 16; ++j) { zz[j] = zbuf[wid][m][j]; mx = fmaxf(mx, zz[j]); }
    float s = 0.0f;
#pragma unroll
    for (int j = 0; j < 16; ++j) { zz[j] = __expf(zz[j] - mx); s += zz[j]; }
    const float inv = 1.0f / s;
    float* g = gates + (size_t)(rowbase + m) * 16;
#pragma unroll
    for (int j = 0; j < 16; ++j) g[j] = zz[j] * inv;
  }
}

// ---------------------------------------------------------------------------
// Kernel 2 (f16-staged inputs): fused GEMM over virtual K' = {src}x{e}x{k}.
// A'[m,k'] = gate[m,src,e] * xh_src[m,k]  (packed f16 scale during LDS fill)
// B tile streams memory -> LDS via global_load_async_to_lds_b128 when the
// toolchain exposes the builtin (ASYNCcnt path); plain ld/st otherwise.
// Block tile 128x256, BK=64, 8 waves (2x4), wave tile 64x64 (4x4 WMMA accs).
// ---------------------------------------------------------------------------
__global__ __launch_bounds__(256) void fused_kernel_f16(
    const _Float16* __restrict__ sph, const _Float16* __restrict__ tph,
    const _Float16* __restrict__ Wsh, const float* __restrict__ bs,
    const _Float16* __restrict__ Wth, const float* __restrict__ bt,
    const float* __restrict__ gates, float* __restrict__ out)
{
  __shared__ __align__(16) _Float16 Alds[BM * LDA];   // 18 KB
  __shared__ __align__(16) _Float16 Blds[BN * LDB];   // 36 KB
  __shared__ float glds[BM * 16];                     // 8 KB

  const int t = threadIdx.x, lane = t & 31, wid = t >> 5;
  const int b0 = blockIdx.y * BM;
  const int f0 = blockIdx.x * BN;
  const int wm = (wid >> 2) * 64;
  const int wn = (wid & 3) * 64;
  const int lrow = lane & 15;
  const int lhi  = lane >> 4;

  for (int i = t; i < BM * 16; i += 256)
    glds[i] = gates[(size_t)(b0 + (i >> 4)) * 16 + (i & 15)];
  __syncthreads();

  v8f acc[4][4] = {};

  for (int src = 0; src < 2; ++src) {
    const _Float16* X = src ? tph : sph;
    const _Float16* W = src ? Wth : Wsh;
    for (int e = 0; e < kE; ++e) {
      for (int k0 = 0; k0 < kS; k0 += BK) {
        // --- B tile (BN cols x BK k): direct memory -> LDS when possible ---
#pragma unroll
        for (int i = 0; i < (BN * BK / 8) / 256; ++i) {   // 8 x 16B / thread
          int slot = t + 256 * i;
          int n  = slot >> 3;                             // 8 h16x8 per column
          int c8 = slot & 7;
          const _Float16* gsrc = W + ((size_t)e * kF + f0 + n) * kS + k0 + c8 * 8;
          _Float16* ldst = &Blds[n * LDB + c8 * 8];
#if defined(USE_ASYNC_LDS)
          __builtin_amdgcn_global_load_async_to_lds_b128((v4i*)gsrc, (v4i*)ldst,
                                                         0, 0);
#else
          *(h16x8*)ldst = *(const h16x8*)gsrc;
#endif
        }
        // --- A tile (BM x BK), packed-f16 scale by per-row gate ---
#pragma unroll
        for (int i = 0; i < (BM * BK / 8) / 256; ++i) {   // 4 x 16B / thread
          int slot = t + 256 * i;
          int row = slot >> 3;
          int c8  = slot & 7;
          _Float16 g = (_Float16)glds[row * 16 + src * 8 + e];
          h16x8 v = *(const h16x8*)(X + (size_t)(b0 + row) * kS + k0 + c8 * 8);
          h16x8 hv;
#pragma unroll
          for (int j = 0; j < 8; ++j) hv[j] = (_Float16)(v[j] * g);
          *(h16x8*)&Alds[row * LDA + c8 * 8] = hv;
        }
#if defined(USE_ASYNC_LDS)
        __builtin_amdgcn_s_wait_asynccnt(0);
#endif
        __syncthreads();
        // --- compute: 2 k-steps x (4 M x 4 N) independent WMMAs per wave ---
#pragma unroll
        for (int kk = 0; kk < BK; kk += 32) {
          v16h af[4], bf[4];
#pragma unroll
          for (int mt = 0; mt < 4; ++mt) {
            const _Float16* p = &Alds[(wm + mt * 16 + lrow) * LDA + kk + lhi * 8];
            af[mt] = cat16(*(const h16x8*)p, *(const h16x8*)(p + 16));
          }
#pragma unroll
          for (int nt = 0; nt < 4; ++nt) {
            const _Float16* p = &Blds[(wn + nt * 16 + lrow) * LDB + kk + lhi * 8];
            bf[nt] = cat16(*(const h16x8*)p, *(const h16x8*)(p + 16));
          }
#pragma unroll
          for (int mt = 0; mt < 4; ++mt)
#pragma unroll
            for (int nt = 0; nt < 4; ++nt)
              acc[mt][nt] = __builtin_amdgcn_wmma_f32_16x16x32_f16(
                  false, af[mt], false, bf[nt], (short)0, acc[mt][nt],
                  false, false);
        }
        __syncthreads();
      }
    }
  }

#pragma unroll
  for (int nt = 0; nt < 4; ++nt) {
    const int f = f0 + wn + nt * 16 + lrow;
    float bsv[kE], btv[kE];
#pragma unroll
    for (int e = 0; e < kE; ++e) { bsv[e] = bs[e * kF + f]; btv[e] = bt[e * kF + f]; }
#pragma unroll
    for (int mt = 0; mt < 4; ++mt) {
#pragma unroll
      for (int v = 0; v < 8; ++v) {
        const int m = wm + mt * 16 + v + lhi * 8;
        float val = acc[mt][nt][v];
#pragma unroll
        for (int e = 0; e < kE; ++e)
          val += glds[m * 16 + e] * bsv[e] + glds[m * 16 + 8 + e] * btv[e];
        out[(size_t)(b0 + m) * kF + f] = val;
      }
    }
  }
}

// ---------------------------------------------------------------------------
// Fallback (workspace too small for f16 staging): f32 inputs, cvt in fill.
// ---------------------------------------------------------------------------
__global__ __launch_bounds__(256) void fused_kernel_f32(
    const float* __restrict__ sp, const float* __restrict__ tp,
    const float* __restrict__ Ws, const float* __restrict__ bs,
    const float* __restrict__ Wt, const float* __restrict__ bt,
    const float* __restrict__ gates, float* __restrict__ out)
{
  __shared__ __align__(16) _Float16 Alds[BM * LDA];
  __shared__ __align__(16) _Float16 Blds[BN * LDB];
  __shared__ float glds[BM * 16];

  const int t = threadIdx.x, lane = t & 31, wid = t >> 5;
  const int b0 = blockIdx.y * BM;
  const int f0 = blockIdx.x * BN;
  const int wm = (wid >> 2) * 64;
  const int wn = (wid & 3) * 64;
  const int lrow = lane & 15;
  const int lhi  = lane >> 4;

  for (int i = t; i < BM * 16; i += 256)
    glds[i] = gates[(size_t)(b0 + (i >> 4)) * 16 + (i & 15)];
  __syncthreads();

  v8f acc[4][4] = {};

  for (int src = 0; src < 2; ++src) {
    const float* X = src ? tp : sp;
    const float* W = src ? Wt : Ws;
    for (int e = 0; e < kE; ++e) {
      for (int k0 = 0; k0 < kS; k0 += BK) {
#pragma unroll
        for (int i = 0; i < (BM * BK / 4) / 256; ++i) {
          int slot = t + 256 * i;
          int row = slot >> 4;
          int c4  = slot & 15;
          float g = glds[row * 16 + src * 8 + e];
          f32x4 v = *(const f32x4*)(X + (size_t)(b0 + row) * kS + k0 + c4 * 4);
          h16x4 hv;
#pragma unroll
          for (int j = 0; j < 4; ++j) hv[j] = (_Float16)(v[j] * g);
          *(h16x4*)&Alds[row * LDA + c4 * 4] = hv;
        }
#pragma unroll
        for (int i = 0; i < (BN * BK / 4) / 256; ++i) {
          int slot = t + 256 * i;
          int n  = slot >> 4;
          int c4 = slot & 15;
          f32x4 v = *(const f32x4*)(W + ((size_t)e * kF + f0 + n) * kS + k0 + c4 * 4);
          h16x4 hv;
#pragma unroll
          for (int j = 0; j < 4; ++j) hv[j] = (_Float16)v[j];
          *(h16x4*)&Blds[n * LDB + c4 * 4] = hv;
        }
        __syncthreads();
#pragma unroll
        for (int kk = 0; kk < BK; kk += 32) {
          v16h af[4], bf[4];
#pragma unroll
          for (int mt = 0; mt < 4; ++mt) {
            const _Float16* p = &Alds[(wm + mt * 16 + lrow) * LDA + kk + lhi * 8];
            af[mt] = cat16(*(const h16x8*)p, *(const h16x8*)(p + 16));
          }
#pragma unroll
          for (int nt = 0; nt < 4; ++nt) {
            const _Float16* p = &Blds[(wn + nt * 16 + lrow) * LDB + kk + lhi * 8];
            bf[nt] = cat16(*(const h16x8*)p, *(const h16x8*)(p + 16));
          }
#pragma unroll
          for (int mt = 0; mt < 4; ++mt)
#pragma unroll
            for (int nt = 0; nt < 4; ++nt)
              acc[mt][nt] = __builtin_amdgcn_wmma_f32_16x16x32_f16(
                  false, af[mt], false, bf[nt], (short)0, acc[mt][nt],
                  false, false);
        }
        __syncthreads();
      }
    }
  }

#pragma unroll
  for (int nt = 0; nt < 4; ++nt) {
    const int f = f0 + wn + nt * 16 + lrow;
    float bsv[kE], btv[kE];
#pragma unroll
    for (int e = 0; e < kE; ++e) { bsv[e] = bs[e * kF + f]; btv[e] = bt[e * kF + f]; }
#pragma unroll
    for (int mt = 0; mt < 4; ++mt) {
#pragma unroll
      for (int v = 0; v < 8; ++v) {
        const int m = wm + mt * 16 + v + lhi * 8;
        float val = acc[mt][nt][v];
#pragma unroll
        for (int e = 0; e < kE; ++e)
          val += glds[m * 16 + e] * bsv[e] + glds[m * 16 + 8 + e] * btv[e];
        out[(size_t)(b0 + m) * kF + f] = val;
      }
    }
  }
}

extern "C" void kernel_launch(void* const* d_in, const int* in_sizes, int n_in,
                              void* d_out, int out_size, void* d_ws, size_t ws_size,
                              hipStream_t stream) {
  (void)in_sizes; (void)n_in; (void)out_size;
  const float* sp  = (const float*)d_in[0];
  const float* tp  = (const float*)d_in[1];
  const float* Ws  = (const float*)d_in[2];
  const float* bs  = (const float*)d_in[3];
  const float* Wt  = (const float*)d_in[4];
  const float* bt  = (const float*)d_in[5];
  const float* Wg1 = (const float*)d_in[6];
  const float* bg1 = (const float*)d_in[7];
  const float* Wg2 = (const float*)d_in[8];
  const float* bg2 = (const float*)d_in[9];
  float* gates = (float*)d_ws;                    // kB*16 f32 = 1 MB
  float* out   = (float*)d_out;

  gates_kernel<<<kB / 128, 256, 0, stream>>>(sp, tp, Wg1, bg1, Wg2, bg2, gates);

  const size_t gate_bytes = (size_t)kB * 16 * sizeof(float);
  const size_t xh_bytes   = (size_t)kB * kS * sizeof(_Float16);       // 32 MB
  const size_t wh_bytes   = (size_t)kE * kF * kS * sizeof(_Float16);  // 32 MB
  const size_t need = gate_bytes + 2 * xh_bytes + 2 * wh_bytes;

  if (ws_size >= need) {
    char* p = (char*)d_ws + gate_bytes;
    _Float16* sph = (_Float16*)p;               p += xh_bytes;
    _Float16* tph = (_Float16*)p;               p += xh_bytes;
    _Float16* Wsh = (_Float16*)p;               p += wh_bytes;
    _Float16* Wth = (_Float16*)p;
    const int n4x = kB * kS / 4;       // 4,194,304
    const int n4w = kE * kF * kS / 4;  // 4,194,304
    cvt_f16_kernel<<<(n4x + 255) / 256, 256, 0, stream>>>(sp, sph, n4x);
    cvt_f16_kernel<<<(n4x + 255) / 256, 256, 0, stream>>>(tp, tph, n4x);
    cvt_f16_kernel<<<(n4w + 255) / 256, 256, 0, stream>>>(Ws, Wsh, n4w);
    cvt_f16_kernel<<<(n4w + 255) / 256, 256, 0, stream>>>(Wt, Wth, n4w);
    fused_kernel_f16<<<dim3(kF / BN, kB / BM), 256, 0, stream>>>(
        sph, tph, Wsh, bs, Wth, bt, gates, out);
  } else {
    fused_kernel_f32<<<dim3(kF / BN, kB / BM), 256, 0, stream>>>(
        sp, tp, Ws, bs, Wt, bt, gates, out);
  }
}
